// ArrangementsDiscreteDecoder_8108898255130
// MI455X (gfx1250) — matrix operand order repaired and checked by reading, not compiled, not run on hardware
//
#include <hip/hip_runtime.h>

// ---------------------------------------------------------------------------
// Model dims (fixed by the reference): H=256, B=16, T_text=512, T_clip=48
// All GEMMs: A [M,K] bf16 row-major (M % 16 == 0, K % 32 == 0),
// weights pre-packed into WMMA B-fragment tile order, N zero-padded to 64.
// ---------------------------------------------------------------------------

typedef __attribute__((ext_vector_type(16))) __bf16 v16bf;
typedef __attribute__((ext_vector_type(8)))  float  v8f;

__device__ __forceinline__ float bf2f(unsigned short s) {
    union { unsigned u; float f; } x; x.u = ((unsigned)s) << 16; return x.f;
}
__device__ __forceinline__ unsigned short f2bf(float f) {
    union { float f; unsigned u; } x; x.f = f;
    unsigned r = x.u + 0x7FFFu + ((x.u >> 16) & 1u);
    return (unsigned short)(r >> 16);
}

struct Frag32B { uint4 lo, hi; };

// A fragment (16x32 bf16): lane holds row m = m0 + (lane&15); its 16 elements
// are two contiguous 8-element K runs at kg and kg+16 (kg = (lane>>4)*8).
__device__ __forceinline__ v16bf load_frag_a(const unsigned short* p) {
    Frag32B t;
    t.lo = *(const uint4*)(p);        // K = kg + 0..7
    t.hi = *(const uint4*)(p + 16);   // K = kg + 16..23
    return __builtin_bit_cast(v16bf, t);
}

// B fragment from pre-packed tiles: fully contiguous 32 bytes per lane.
__device__ __forceinline__ v16bf load_frag_packed(const unsigned short* p) {
    Frag32B t;
    t.lo = *(const uint4*)(p);
    t.hi = *(const uint4*)(p + 8);
    return __builtin_bit_cast(v16bf, t);
}

// ---------------------------------------------------------------------------
// Weight pre-pack: W[N,K] f32 -> bf16 tiles [nt][kt][lane(=N)][16(=K)],
// N zero-padded to Npad (multiple of 64). Makes the GEMM a pure b128 stream.
// ---------------------------------------------------------------------------
__global__ void pack_weights(const float* __restrict__ W, unsigned short* __restrict__ dst,
                             int N, int K, int Npad) {
    const int Kt = K >> 5;
    long total = (long)(Npad >> 4) * Kt * 512;
    for (long i = blockIdx.x * (long)blockDim.x + threadIdx.x; i < total;
         i += (long)gridDim.x * blockDim.x) {
        int  e    = (int)(i & 15);
        int  l    = (int)((i >> 4) & 31);
        long tile = i >> 9;
        int  kt   = (int)(tile % Kt);
        int  nt   = (int)(tile / Kt);
        int n = nt * 16 + (l & 15);
        int k = kt * 32 + (l >> 4) * 8 + (e & 7) + (e >> 3) * 16;
        float v = (n < N) ? W[(long)n * K + k] : 0.0f;
        dst[i] = f2bf(v);
    }
}

// ---------------------------------------------------------------------------
// GEMM: C[M,N] = A[M,K] * W^T + bias. Block = 256 thr = 8 waves.
// Block tile 128(M) x 64(N); wave = 16M x 64N (4 WMMA per K-step).
// ---------------------------------------------------------------------------
__global__ void wmma_gemm_bias(const unsigned short* __restrict__ A,
                               const unsigned short* __restrict__ Wp,
                               const float* __restrict__ bias,
                               float* __restrict__ C,
                               int M, int N, int K) {
    const int lane = threadIdx.x & 31;
    const int wid  = threadIdx.x >> 5;
    const int m0   = blockIdx.x * 128 + wid * 16;
    const int nt0  = blockIdx.y * 4;
    const int Kt   = K >> 5;

    const unsigned short* a_row = A + (long)(m0 + (lane & 15)) * K + ((lane >> 4) * 8);
    const unsigned short* b0    = Wp + ((long)nt0 * Kt * 32 + lane) * 16;

    v8f acc[4] = {};
    for (int kt = 0; kt < Kt; ++kt) {
        __builtin_prefetch(a_row + kt * 32 + 128, 0, 1);
        v16bf a = load_frag_a(a_row + kt * 32);
#pragma unroll
        for (int j = 0; j < 4; ++j) {
            const unsigned short* bp = b0 + ((long)j * Kt + kt) * 512;
            __builtin_prefetch(bp + 512, 0, 1);
            v16bf b = load_frag_packed(bp);
            acc[j] = __builtin_amdgcn_wmma_f32_16x16x32_bf16(
                false, a, false, b, (short)0, acc[j], false, false);
        }
    }
    const int mrow = m0 + 8 * (lane >> 4);
    const int nc   = lane & 15;
#pragma unroll
    for (int j = 0; j < 4; ++j) {
        int n = (nt0 + j) * 16 + nc;
        if (n >= N) continue;                 // only ragged-N epilogue is guarded
        float bb = bias ? bias[n] : 0.0f;
#pragma unroll
        for (int v = 0; v < 8; ++v)
            C[(long)(mrow + v) * N + n] = acc[j][v] + bb;
    }
}

// ---------------------------------------------------------------------------
// Persistent GRU recurrences, all four directions in ONE launch:
//   block 0/1 = text fwd/bwd (T=Tt), block 2/3 = clip fwd/bwd (T=Tc).
// h [16,256] in LDS (f32+bf16); gh = h @ Wh^T via WMMA (8 waves x 6 Nt x 8 Kt).
// The per-step gi slice (16x768 f32 = 48KB) is double-buffered in LDS via
// GLOBAL_LOAD_ASYNC_TO_LDS_B128, issued before the WMMA phase so the copy
// overlaps matrix work; S_WAIT_ASYNCCNT 12 releases the current buffer while
// next-step prefetches (12 async ops/wave) remain in flight.
// LDS: hf 16K | hb 8K | gh 48K | giA 48K | giB 48K = 168KB (1 WG per WGP).
// ---------------------------------------------------------------------------
#define GRU_LDS_HF   0
#define GRU_LDS_HB   (16 * 256 * 4)
#define GRU_LDS_GH   (GRU_LDS_HB + 16 * 256 * 2)
#define GRU_LDS_GIA  (GRU_LDS_GH + 16 * 768 * 4)
#define GRU_LDS_GIB  (GRU_LDS_GIA + 16 * 768 * 4)
#define GRU_LDS_SIZE (GRU_LDS_GIB + 16 * 768 * 4)

__device__ __forceinline__ void gru_prefetch_gi(const float* gi, int T, int t,
                                                unsigned buf_off, int tid) {
#pragma unroll
    for (int j = 0; j < 12; ++j) {
        int c  = tid + 256 * j;          // 3072 chunks of 16B = 16 rows x 3KB
        int b_ = c / 192;
        int cc = c - b_ * 192;
        unsigned long long ga =
            (unsigned long long)(gi + ((long)b_ * T + t) * 768 + (long)cc * 4);
        unsigned loff = buf_off + (unsigned)c * 16;
        asm volatile("global_load_async_to_lds_b128 %0, %1, off"
                     :: "v"(loff), "v"(ga) : "memory");
    }
}

__global__ void gru_recurrent(const float* __restrict__ gi_tf, const float* __restrict__ gi_tb,
                              const float* __restrict__ gi_cf, const float* __restrict__ gi_cb,
                              const unsigned short* __restrict__ Wtf, const unsigned short* __restrict__ Wtb,
                              const unsigned short* __restrict__ Wcf, const unsigned short* __restrict__ Wcb,
                              const float* __restrict__ bhtf, const float* __restrict__ bhtb,
                              const float* __restrict__ bhcf, const float* __restrict__ bhcb,
                              unsigned short* __restrict__ enc_t, unsigned short* __restrict__ enc_c,
                              int Tt, int Tc) {
    extern __shared__ char smem[];
    float*          h_f = (float*)(smem + GRU_LDS_HF);
    unsigned short* h_b = (unsigned short*)(smem + GRU_LDS_HB);
    float*          gh  = (float*)(smem + GRU_LDS_GH);
    float*          giA = (float*)(smem + GRU_LDS_GIA);
    float*          giB = (float*)(smem + GRU_LDS_GIB);

    const int id  = blockIdx.x;          // 0,1: text f/b ; 2,3: clip f/b
    const int dir = id & 1;
    const bool is_text = (id < 2);
    const float* gi = is_text ? (dir ? gi_tb : gi_tf) : (dir ? gi_cb : gi_cf);
    const unsigned short* Whp = is_text ? (dir ? Wtb : Wtf) : (dir ? Wcb : Wcf);
    const float* bh = is_text ? (dir ? bhtb : bhtf) : (dir ? bhcb : bhcf);
    unsigned short* enc = is_text ? enc_t : enc_c;
    const int T = is_text ? Tt : Tc;

    const int tid = threadIdx.x, lane = tid & 31, wid = tid >> 5;

    for (int i = tid; i < 16 * 256; i += 256) { h_f[i] = 0.0f; h_b[i] = 0; }

    // prologue: async-stage gi for step 0 into buffer A
    gru_prefetch_gi(gi, T, dir ? (T - 1) : 0, GRU_LDS_GIA, tid);
    __syncthreads();

    const unsigned short* a_row  = h_b + (lane & 15) * 256 + ((lane >> 4) * 8);
    const unsigned short* b_base = Whp + ((long)(wid * 6) * 8 * 32 + lane) * 16;

    for (int step = 0; step < T; ++step) {
        int t = dir ? (T - 1 - step) : step;

        // kick off next step's gi copy (overlaps with WMMA below)
        bool more = (step + 1 < T);
        if (more) {
            int tn = dir ? (T - 2 - step) : (step + 1);
            gru_prefetch_gi(gi, T, tn, ((step + 1) & 1) ? GRU_LDS_GIB : GRU_LDS_GIA, tid);
        }

        // phase 1: gh = h @ Wh^T
        v8f acc[6] = {};
        for (int kt = 0; kt < 8; ++kt) {
            v16bf a = load_frag_a(a_row + kt * 32);
#pragma unroll
            for (int j = 0; j < 6; ++j) {
                v16bf b = load_frag_packed(b_base + ((long)j * 8 + kt) * 512);
                acc[j] = __builtin_amdgcn_wmma_f32_16x16x32_bf16(
                    false, a, false, b, (short)0, acc[j], false, false);
            }
        }
        const int mrow = 8 * (lane >> 4);
        const int nc   = lane & 15;
#pragma unroll
        for (int j = 0; j < 6; ++j) {
            int n = (wid * 6 + j) * 16 + nc;
#pragma unroll
            for (int v = 0; v < 8; ++v) gh[(mrow + v) * 768 + n] = acc[j][v];
        }
        // release current gi buffer (12 next-step async ops may stay in flight)
        if (more) asm volatile("s_wait_asynccnt 0xc" ::: "memory");
        else      asm volatile("s_wait_asynccnt 0x0" ::: "memory");
        __syncthreads();

        // phase 2: gates (PyTorch convention), update h, emit bf16 output
        const float* gi_cur = (step & 1) ? giB : giA;
        for (int i = tid; i < 16 * 256; i += 256) {
            int b_ = i >> 8, hh = i & 255;
            float gir = gi_cur[b_ * 768 + hh];
            float giz = gi_cur[b_ * 768 + 256 + hh];
            float gin = gi_cur[b_ * 768 + 512 + hh];
            float ghr = gh[b_ * 768 + hh]       + bh[hh];
            float ghz = gh[b_ * 768 + 256 + hh] + bh[256 + hh];
            float ghn = gh[b_ * 768 + 512 + hh] + bh[512 + hh];
            float r  = 1.0f / (1.0f + __expf(-(gir + ghr)));
            float z  = 1.0f / (1.0f + __expf(-(giz + ghz)));
            float nn = tanhf(gin + r * ghn);
            float hp = h_f[i];
            float hn = (1.0f - z) * nn + z * hp;
            h_f[i] = hn;
            h_b[i] = f2bf(hn);
            enc[((long)b_ * T + t) * 512 + dir * 256 + hh] = f2bf(hn);
        }
        __syncthreads();
    }
}

// ---------------------------------------------------------------------------
// Embedding gather + bf16 cast, vectorized: float4 in, 4 packed bf16 out.
// ---------------------------------------------------------------------------
__global__ void embed_cast(const int* __restrict__ idx, const float* __restrict__ emb,
                           unsigned short* __restrict__ out, long rows) {
    long total4 = rows * 64;   // H=256 -> 64 float4 groups per row
    for (long i = blockIdx.x * (long)blockDim.x + threadIdx.x; i < total4;
         i += (long)gridDim.x * blockDim.x) {
        long row = i >> 6;
        int  c4  = (int)(i & 63);
        float4 v = *(const float4*)(emb + (long)idx[row] * 256 + c4 * 4);
        uint2 pk;
        pk.x = (unsigned)f2bf(v.x) | ((unsigned)f2bf(v.y) << 16);
        pk.y = (unsigned)f2bf(v.z) | ((unsigned)f2bf(v.w) << 16);
        *(uint2*)(out + row * 256 + (long)c4 * 4) = pk;
    }
}

// ---------------------------------------------------------------------------
// Fused Bahdanau attention: per (b,c) block; the [B,Tc,Tt,H] tanh tensor is
// never materialized. Score phase: each lane owns 8 contiguous H values
// (two float4 loads); context phase: 2 packed bf16 columns per thread.
// ---------------------------------------------------------------------------
__global__ void attention_fused(const float* __restrict__ o_v,   // [B,Tt,256]
                                const float* __restrict__ o_h,   // [B*Tc,256]
                                const float* __restrict__ Wu,    // [256]
                                const unsigned short* __restrict__ enc_t, // [B,Tt,512]
                                float* __restrict__ attn,        // [B*Tc,512]
                                int Tt, int Tc) {
    __shared__ float oh_s[256];
    __shared__ float wu_s[256];
    __shared__ float sc[512];
    __shared__ float red[256];
    int r = blockIdx.x;           // b*Tc + c
    int b = r / Tc;
    int tid = threadIdx.x, lane = tid & 31, wid = tid >> 5;

    oh_s[tid] = o_h[(long)r * 256 + tid];
    wu_s[tid] = Wu[tid];
    __syncthreads();

    for (int t = wid; t < Tt; t += 8) {
        const float4* ov4 = (const float4*)(o_v + ((long)b * Tt + t) * 256) + lane * 2;
        float4 a0 = ov4[0], a1 = ov4[1];
        int hb = lane * 8;
        float p;
        p  = wu_s[hb + 0] * tanhf(a0.x + oh_s[hb + 0]);
        p += wu_s[hb + 1] * tanhf(a0.y + oh_s[hb + 1]);
        p += wu_s[hb + 2] * tanhf(a0.z + oh_s[hb + 2]);
        p += wu_s[hb + 3] * tanhf(a0.w + oh_s[hb + 3]);
        p += wu_s[hb + 4] * tanhf(a1.x + oh_s[hb + 4]);
        p += wu_s[hb + 5] * tanhf(a1.y + oh_s[hb + 5]);
        p += wu_s[hb + 6] * tanhf(a1.z + oh_s[hb + 6]);
        p += wu_s[hb + 7] * tanhf(a1.w + oh_s[hb + 7]);
#pragma unroll
        for (int off = 16; off > 0; off >>= 1) p += __shfl_xor(p, off, 32);
        if (lane == 0) sc[t] = p;
    }
    __syncthreads();

    float lm = -3.4e38f;
    for (int t = tid; t < Tt; t += 256) lm = fmaxf(lm, sc[t]);
    red[tid] = lm; __syncthreads();
    for (int s = 128; s > 0; s >>= 1) { if (tid < s) red[tid] = fmaxf(red[tid], red[tid + s]); __syncthreads(); }
    float mx = red[0]; __syncthreads();
    float ls = 0.0f;
    for (int t = tid; t < Tt; t += 256) { float e = __expf(sc[t] - mx); sc[t] = e; ls += e; }
    red[tid] = ls; __syncthreads();
    for (int s = 128; s > 0; s >>= 1) { if (tid < s) red[tid] += red[tid + s]; __syncthreads(); }
    float inv = 1.0f / red[0]; __syncthreads();

    // context: 2 adjacent columns per thread, packed bf16 pair per load
    {
        int d0 = tid * 2;
        const unsigned* e0 = (const unsigned*)(enc_t + ((long)b * Tt) * 512 + d0);
        float acc0 = 0.0f, acc1 = 0.0f;
        for (int t = 0; t < Tt; ++t) {
            unsigned u = e0[(long)t * 256];
            acc0 += sc[t] * bf2f((unsigned short)(u & 0xFFFF));
            acc1 += sc[t] * bf2f((unsigned short)(u >> 16));
        }
        float2 o; o.x = acc0 * inv; o.y = acc1 * inv;
        *(float2*)(attn + (long)r * 512 + d0) = o;
    }
}

// hidden[768,1024] = concat(clip_enc bf16 [768,512], bf16(attn) [768,512])
__global__ void build_hidden(const unsigned short* __restrict__ enc_c,
                             const float* __restrict__ attn,
                             unsigned short* __restrict__ hidden, long rows) {
    for (long i = blockIdx.x * (long)blockDim.x + threadIdx.x; i < rows * 512;
         i += (long)gridDim.x * blockDim.x) {
        long row = i >> 9; int col = (int)(i & 511);
        hidden[row * 1024 + col]       = enc_c[i];
        hidden[row * 1024 + 512 + col] = f2bf(attn[i]);
    }
}

// log_softmax over last dim for the three heads; writes final d_out
__global__ void head_logsoftmax(const float* __restrict__ xl, const float* __restrict__ yl,
                                const float* __restrict__ ol, float* __restrict__ out, int rows) {
    __shared__ float red[128];
    int r = blockIdx.x, tid = threadIdx.x;
    const float* srcs[3] = { xl, yl, ol };
    const int    ns[3]   = { 99, 99, 7 };
    long bases[3];
    bases[0] = 0; bases[1] = (long)rows * 99; bases[2] = (long)rows * 198;
    for (int hsel = 0; hsel < 3; ++hsel) {
        int n = ns[hsel];
        const float* s = srcs[hsel] + (long)r * n;
        float lm = -3.4e38f;
        for (int i = tid; i < n; i += 128) lm = fmaxf(lm, s[i]);
        red[tid] = lm; __syncthreads();
        for (int st = 64; st > 0; st >>= 1) { if (tid < st) red[tid] = fmaxf(red[tid], red[tid + st]); __syncthreads(); }
        float mx = red[0]; __syncthreads();
        float ls = 0.0f;
        for (int i = tid; i < n; i += 128) ls += __expf(s[i] - mx);
        red[tid] = ls; __syncthreads();
        for (int st = 64; st > 0; st >>= 1) { if (tid < st) red[tid] += red[tid + st]; __syncthreads(); }
        float lse = mx + __logf(red[0]); __syncthreads();
        for (int i = tid; i < n; i += 128) out[bases[hsel] + (long)r * n + i] = s[i] - lse;
    }
}

// ---------------------------------------------------------------------------
extern "C" void kernel_launch(void* const* d_in, const int* in_sizes, int n_in,
                              void* d_out, int out_size, void* d_ws, size_t ws_size,
                              hipStream_t stream) {
    const int*   text_inds = (const int*)d_in[0];
    const int*   clip_inds = (const int*)d_in[1];
    const float* text_emb  = (const float*)d_in[2];
    const float* clip_emb  = (const float*)d_in[3];
    const float* tWi_f = (const float*)d_in[4],  *tWh_f = (const float*)d_in[5];
    const float* tbi_f = (const float*)d_in[6],  *tbh_f = (const float*)d_in[7];
    const float* tWi_b = (const float*)d_in[8],  *tWh_b = (const float*)d_in[9];
    const float* tbi_b = (const float*)d_in[10], *tbh_b = (const float*)d_in[11];
    const float* cWi_f = (const float*)d_in[12], *cWh_f = (const float*)d_in[13];
    const float* cbi_f = (const float*)d_in[14], *cbh_f = (const float*)d_in[15];
    const float* cWi_b = (const float*)d_in[16], *cWh_b = (const float*)d_in[17];
    const float* cbi_b = (const float*)d_in[18], *cbh_b = (const float*)d_in[19];
    const float* Wv  = (const float*)d_in[20], *bv  = (const float*)d_in[21];
    const float* Wha = (const float*)d_in[22], *bha = (const float*)d_in[23];
    const float* Wu  = (const float*)d_in[24];
    const float* Wx  = (const float*)d_in[25], *bx = (const float*)d_in[26];
    const float* Wy  = (const float*)d_in[27], *by = (const float*)d_in[28];
    const float* Wo  = (const float*)d_in[29], *bo = (const float*)d_in[30];

    const int B = 16, Tt = 512, Tc = 48;

    char* wsp = (char*)d_ws;
    auto alloc = [&](size_t bytes) -> char* {
        char* p = wsp; wsp += (bytes + 255) & ~(size_t)255; return p;
    };

    unsigned short* xt = (unsigned short*)alloc((size_t)B * Tt * 256 * 2);
    unsigned short* xc = (unsigned short*)alloc((size_t)B * Tc * 256 * 2);
    unsigned short* tWi_f_p = (unsigned short*)alloc((size_t)768 * 256 * 2);
    unsigned short* tWh_f_p = (unsigned short*)alloc((size_t)768 * 256 * 2);
    unsigned short* tWi_b_p = (unsigned short*)alloc((size_t)768 * 256 * 2);
    unsigned short* tWh_b_p = (unsigned short*)alloc((size_t)768 * 256 * 2);
    unsigned short* cWi_f_p = (unsigned short*)alloc((size_t)768 * 256 * 2);
    unsigned short* cWh_f_p = (unsigned short*)alloc((size_t)768 * 256 * 2);
    unsigned short* cWi_b_p = (unsigned short*)alloc((size_t)768 * 256 * 2);
    unsigned short* cWh_b_p = (unsigned short*)alloc((size_t)768 * 256 * 2);
    unsigned short* Wv_p  = (unsigned short*)alloc((size_t)256 * 512 * 2);
    unsigned short* Wha_p = (unsigned short*)alloc((size_t)256 * 512 * 2);
    unsigned short* Wx_p  = (unsigned short*)alloc((size_t)128 * 1024 * 2);
    unsigned short* Wy_p  = (unsigned short*)alloc((size_t)128 * 1024 * 2);
    unsigned short* Wo_p  = (unsigned short*)alloc((size_t)64 * 1024 * 2);
    float* gi_tf = (float*)alloc((size_t)B * Tt * 768 * 4);
    float* gi_tb = (float*)alloc((size_t)B * Tt * 768 * 4);
    float* gi_cf = (float*)alloc((size_t)B * Tc * 768 * 4);
    float* gi_cb = (float*)alloc((size_t)B * Tc * 768 * 4);
    unsigned short* enc_t = (unsigned short*)alloc((size_t)B * Tt * 512 * 2);
    unsigned short* enc_c = (unsigned short*)alloc((size_t)B * Tc * 512 * 2);
    float* o_v  = (float*)alloc((size_t)B * Tt * 256 * 4);
    float* o_hh = (float*)alloc((size_t)B * Tc * 256 * 4);
    float* attn = (float*)alloc((size_t)B * Tc * 512 * 4);
    unsigned short* hidden = (unsigned short*)alloc((size_t)B * Tc * 1024 * 2);
    float* xl = (float*)alloc((size_t)B * Tc * 99 * 4);
    float* yl = (float*)alloc((size_t)B * Tc * 99 * 4);
    float* ol = (float*)alloc((size_t)B * Tc * 7 * 4);

    // 1) pack all GEMM weights into fragment tile order (one-time)
    struct { const float* s; unsigned short* d; int N, K, Npad; } packs[] = {
        { tWi_f, tWi_f_p, 768, 256, 768 }, { tWh_f, tWh_f_p, 768, 256, 768 },
        { tWi_b, tWi_b_p, 768, 256, 768 }, { tWh_b, tWh_b_p, 768, 256, 768 },
        { cWi_f, cWi_f_p, 768, 256, 768 }, { cWh_f, cWh_f_p, 768, 256, 768 },
        { cWi_b, cWi_b_p, 768, 256, 768 }, { cWh_b, cWh_b_p, 768, 256, 768 },
        { Wv,  Wv_p,  256, 512, 256 }, { Wha, Wha_p, 256, 512, 256 },
        { Wx,  Wx_p,  99, 1024, 128 }, { Wy,  Wy_p,  99, 1024, 128 },
        { Wo,  Wo_p,  7,  1024, 64  },
    };
    for (auto& p : packs) {
        long total = (long)(p.Npad >> 4) * (p.K >> 5) * 512;
        pack_weights<<<dim3((unsigned)((total + 255) / 256)), dim3(256), 0, stream>>>(
            p.s, p.d, p.N, p.K, p.Npad);
    }

    // 2) embedding gather + cast
    embed_cast<<<dim3(1024), dim3(256), 0, stream>>>(text_inds, text_emb, xt, (long)B * Tt);
    embed_cast<<<dim3(192),  dim3(256), 0, stream>>>(clip_inds, clip_emb, xc, (long)B * Tc);

    // 3) input projections gi = x @ Wi^T + bi   (M x 768, K=256)
    wmma_gemm_bias<<<dim3(64, 12), dim3(256), 0, stream>>>(xt, tWi_f_p, tbi_f, gi_tf, B * Tt, 768, 256);
    wmma_gemm_bias<<<dim3(64, 12), dim3(256), 0, stream>>>(xt, tWi_b_p, tbi_b, gi_tb, B * Tt, 768, 256);
    wmma_gemm_bias<<<dim3(6, 12),  dim3(256), 0, stream>>>(xc, cWi_f_p, cbi_f, gi_cf, B * Tc, 768, 256);
    wmma_gemm_bias<<<dim3(6, 12),  dim3(256), 0, stream>>>(xc, cWi_b_p, cbi_b, gi_cb, B * Tc, 768, 256);

    // 4) all four GRU directions in one launch (text & clip run concurrently)
    gru_recurrent<<<dim3(4), dim3(256), GRU_LDS_SIZE, stream>>>(
        gi_tf, gi_tb, gi_cf, gi_cb,
        tWh_f_p, tWh_b_p, cWh_f_p, cWh_b_p,
        tbh_f, tbh_b, cbh_f, cbh_b,
        enc_t, enc_c, Tt, Tc);

    // 5) attention projections (K=512 -> N=256)
    wmma_gemm_bias<<<dim3(64, 4), dim3(256), 0, stream>>>(enc_t, Wv_p,  bv,  o_v,  B * Tt, 256, 512);
    wmma_gemm_bias<<<dim3(6, 4),  dim3(256), 0, stream>>>(enc_c, Wha_p, bha, o_hh, B * Tc, 256, 512);

    // 6) fused score + softmax + context (one WG per (b,c))
    attention_fused<<<dim3(B * Tc), dim3(256), 0, stream>>>(o_v, o_hh, Wu, enc_t, attn, Tt, Tc);

    // 7) hidden = concat(clip_enc, attn) as bf16 [768,1024]
    build_hidden<<<dim3(1536), dim3(256), 0, stream>>>(enc_c, attn, hidden, (long)B * Tc);

    // 8) heads (K=1024; Npad 128/128/64 -> grid.y 2/2/1)
    wmma_gemm_bias<<<dim3(6, 2), dim3(256), 0, stream>>>(hidden, Wx_p, bx, xl, B * Tc, 99, 1024);
    wmma_gemm_bias<<<dim3(6, 2), dim3(256), 0, stream>>>(hidden, Wy_p, by, yl, B * Tc, 99, 1024);
    wmma_gemm_bias<<<dim3(6, 1), dim3(256), 0, stream>>>(hidden, Wo_p, bo, ol, B * Tc, 7, 1024);

    // 9) log_softmax -> d_out (x[768,99] | y[768,99] | o[768,7])
    head_logsoftmax<<<dim3(B * Tc), dim3(128), 0, stream>>>(xl, yl, ol, (float*)d_out, B * Tc);
}
